// _MultiHeadAttention_48911087567283
// MI455X (gfx1250) — matrix operand order repaired and checked by reading, not compiled, not run on hardware
//
#include <hip/hip_runtime.h>
#include <hip/hip_bf16.h>
#include <math.h>

// MHA forward for MI455X (gfx1250, wave32, WMMA + async LDS DMA).
// Shapes: B=2, S=2048, D=1024, H=16, Dh=64. SCALE = 1/8.
// d_out = [ out (B*S*D fp32) | attn (B*H*S*S fp32) ].

typedef __attribute__((ext_vector_type(16))) _Float16 v16h;
typedef __attribute__((ext_vector_type(8)))  float    v8f;

#define S_LEN   2048
#define D_MODEL 1024
#define NHEADS  16
#define HEAD_D  64
#define MROWS   4096            // B*S
#define ATTN_SCALE 0.125f

// ---------------------------------------------------------------------------
// GEMM: out[M,N] = X[M,K] @ W[N,K]^T + bias[N], M=4096, N=K=1024.
// Block = 128 threads (4 waves). Each wave computes one 16x16 tile.
// ---------------------------------------------------------------------------
template <typename OutT>
__global__ __launch_bounds__(128)
void gemm_bias_bT(const float* __restrict__ X,
                  const float* __restrict__ W,
                  const float* __restrict__ bias,
                  OutT* __restrict__ out)
{
    const int tid  = threadIdx.x;
    const int wave = tid >> 5;
    const int lane = tid & 31;
    const int l16  = lane & 15;
    const int hs   = lane >> 4;          // half-wave select
    const int c0   = hs * 16;            // A-fragment K contiguous base

    const int mt = blockIdx.x;
    const int nt = blockIdx.y * 4 + wave;

    const int arow = mt * 16 + l16;
    const int ncol = nt * 16 + l16;

    const float* __restrict__ Arow = X + (size_t)arow * D_MODEL;
    const float* __restrict__ Wrow = W + (size_t)ncol * D_MODEL;

    v8f acc = {};
    for (int kk = 0; kk < D_MODEL; kk += 32) {
        __builtin_prefetch(Arow + kk + 32, 0, 1);   // global_prefetch_b8

        v16h a, bf;
#pragma unroll
        for (int j = 0; j < 8; ++j) {
            a[j]     = (_Float16)Arow[kk + c0 + j];
            a[j + 8] = (_Float16)Arow[kk + c0 + 8 + j];
        }
        const float* wp = Wrow + kk + hs * 16;
#pragma unroll
        for (int e = 0; e < 16; ++e) bf[e] = (_Float16)wp[e];

        acc = __builtin_amdgcn_wmma_f32_16x16x32_f16(
            false, a, false, bf, (short)0, acc, false, false);
    }

    const float bv = bias[ncol];
#pragma unroll
    for (int v = 0; v < 8; ++v) {
        const int orow = mt * 16 + v + 8 * hs;
        out[(size_t)orow * D_MODEL + ncol] = (OutT)(acc[v] + bv);
    }
}

// ---------------------------------------------------------------------------
// Async DMA of one 32x64 f16 V tile (4 KB) into LDS, 128 threads cooperate:
// thread t moves 32 bytes (two b128 chunks) with GLOBAL_LOAD_ASYNC_TO_LDS.
// The instruction's IOFFSET applies to BOTH the LDS and global address, so
// one base pair + offset:16 covers the second chunk.
// ---------------------------------------------------------------------------
__device__ __forceinline__ void issue_vtile_load(const _Float16* gsrc,
                                                 _Float16* lds_dst)
{
    unsigned laddr = (unsigned)(unsigned long long)(const void*)lds_dst;
    unsigned long long gaddr = (unsigned long long)gsrc;
    asm volatile("global_load_async_to_lds_b128 %0, %1, off"
                 :: "v"(laddr), "v"(gaddr) : "memory");
    asm volatile("global_load_async_to_lds_b128 %0, %1, off offset:16"
                 :: "v"(laddr), "v"(gaddr) : "memory");
}

// ---------------------------------------------------------------------------
// Attention: one block (128 thr / 4 waves) per (batch, head, 16-row Q tile).
// 16x2048 fp32 score strip + 16x2048 f16 normalized copy live in LDS
// (128 KB + 64 KB of the 320 KB WGP LDS); attn hits HBM exactly once.
// PV stage double-buffers 32x64 V tiles into LDS via async DMA.
// ---------------------------------------------------------------------------
__global__ __launch_bounds__(128)
void attn_kernel(const _Float16* __restrict__ qs,
                 const _Float16* __restrict__ ks,
                 const _Float16* __restrict__ vs,
                 float* __restrict__ attn_g,     // [B,H,S,S]
                 float* __restrict__ head_out)   // [B,S,D] merged heads
{
    __shared__ float sc[16 * S_LEN];                       // 128 KB
    __shared__ _Float16 scf16[16 * S_LEN];                 //  64 KB
    __shared__ __align__(16) _Float16 vtile[2][32 * 64];   //   8 KB
    __shared__ float red[128];
    __shared__ float rowstat[16];

    const int tid  = threadIdx.x;
    const int wave = tid >> 5;
    const int lane = tid & 31;
    const int l16  = lane & 15;
    const int hs   = lane >> 4;
    const int c0   = hs * 16;

    const int qt = blockIdx.x & 127;
    const int h  = (blockIdx.x >> 7) & 15;
    const int b  = blockIdx.x >> 11;

    // ---- Q fragments for this 16-row tile (Dh=64 -> two K=32 steps) ----
    const int qrow = qt * 16 + l16;
    const size_t qbase = ((size_t)b * S_LEN + qrow) * D_MODEL + h * HEAD_D;
    v16h aq0, aq1;
#pragma unroll
    for (int j = 0; j < 8; ++j) {
        aq0[j]     = qs[qbase + c0 + j];
        aq0[j + 8] = qs[qbase + c0 + 8 + j];
        aq1[j]     = qs[qbase + 32 + c0 + j];
        aq1[j + 8] = qs[qbase + 32 + c0 + 8 + j];
    }

    // ---- scores: waves stripe over the 128 key tiles ----
    for (int kt = wave; kt < S_LEN / 16; kt += 4) {
        const int kcol = kt * 16 + l16;
        const size_t kb = ((size_t)b * S_LEN + kcol) * D_MODEL + h * HEAD_D + hs * 16;
        v16h b0, b1;
#pragma unroll
        for (int e = 0; e < 16; ++e) {
            b0[e] = ks[kb + e];
            b1[e] = ks[kb + 32 + e];
        }
        v8f c = {};
        c = __builtin_amdgcn_wmma_f32_16x16x32_f16(false, aq0, false, b0, (short)0, c, false, false);
        c = __builtin_amdgcn_wmma_f32_16x16x32_f16(false, aq1, false, b1, (short)0, c, false, false);
#pragma unroll
        for (int v = 0; v < 8; ++v)
            sc[(v + 8 * hs) * S_LEN + kt * 16 + l16] = c[v] * ATTN_SCALE;
    }
    __syncthreads();

    // ---- softmax: 8 threads x 256 elems per row ----
    const int r   = tid >> 3;
    const int seg = tid & 7;
    float* rowp = &sc[r * S_LEN + seg * 256];

    float m = -INFINITY;
    for (int i = 0; i < 256; ++i) m = fmaxf(m, rowp[i]);
    red[tid] = m;
    __syncthreads();
    if (tid < 16) {
        float mm = red[tid * 8];
        for (int j = 1; j < 8; ++j) mm = fmaxf(mm, red[tid * 8 + j]);
        rowstat[tid] = mm;
    }
    __syncthreads();
    const float rmax = rowstat[r];
    float s = 0.f;
    for (int i = 0; i < 256; ++i) {
        const float e = __expf(rowp[i] - rmax);
        rowp[i] = e;
        s += e;
    }
    red[tid] = s;
    __syncthreads();
    if (tid < 16) {
        float ss = red[tid * 8];
        for (int j = 1; j < 8; ++j) ss += red[tid * 8 + j];
        rowstat[tid] = ss;
    }
    __syncthreads();

    // ---- normalize: write attn to HBM once + f16 copy to LDS for PV ----
    const float inv = 1.0f / rowstat[r];
    _Float16* f16row = &scf16[r * S_LEN + seg * 256];
    float* ag = attn_g + (((size_t)(b * NHEADS + h) * S_LEN) + (qt * 16 + r)) * S_LEN
                       + seg * 256;
    for (int i = 0; i < 256; ++i) {
        const float val = rowp[i] * inv;
        f16row[i] = (_Float16)val;
        ag[i]     = val;
    }
    __syncthreads();

    // ---- PV with double-buffered async V-tile DMA ----
    const int t_row = tid >> 2;          // 0..31
    const int t_col = (tid & 3) * 16;    // 0,16,32,48
    const int w = wave;

    const _Float16* vbase = vs + ((size_t)b * S_LEN) * D_MODEL + h * HEAD_D;

    issue_vtile_load(vbase + (size_t)(0 * 32 + t_row) * D_MODEL + t_col,
                     &vtile[0][t_row * 64 + t_col]);

    v8f o = {};
    for (int kt = 0; kt < S_LEN / 32; ++kt) {
        const int buf = kt & 1;
        if (kt + 1 < S_LEN / 32) {
            issue_vtile_load(vbase + (size_t)((kt + 1) * 32 + t_row) * D_MODEL + t_col,
                             &vtile[buf ^ 1][t_row * 64 + t_col]);
            asm volatile("s_wait_asynccnt 2" ::: "memory");  // current tile done
        } else {
            asm volatile("s_wait_asynccnt 0" ::: "memory");
        }
        __syncthreads();

        v16h a, bvf;
        const _Float16* ap = &scf16[l16 * S_LEN + kt * 32 + c0];
#pragma unroll
        for (int j = 0; j < 8; ++j) {
            a[j]     = ap[j];        // one contiguous 32B run -> ds_load_b128 x2
            a[j + 8] = ap[8 + j];
        }
#pragma unroll
        for (int e = 0; e < 16; ++e)
            bvf[e] = vtile[buf][(hs * 16 + e) * 64 + w * 16 + l16];

        o = __builtin_amdgcn_wmma_f32_16x16x32_f16(false, a, false, bvf, (short)0, o, false, false);
        __syncthreads();
    }

#pragma unroll
    for (int v = 0; v < 8; ++v) {
        const size_t orow = (size_t)b * S_LEN + qt * 16 + v + 8 * hs;
        head_out[orow * D_MODEL + h * HEAD_D + w * 16 + l16] = o[v];
    }
}

// ---------------------------------------------------------------------------
extern "C" void kernel_launch(void* const* d_in, const int* in_sizes, int n_in,
                              void* d_out, int out_size, void* d_ws, size_t ws_size,
                              hipStream_t stream) {
    const float* q  = (const float*)d_in[0];
    const float* k  = (const float*)d_in[1];
    const float* v  = (const float*)d_in[2];
    const float* Wq = (const float*)d_in[3];
    const float* bq = (const float*)d_in[4];
    const float* Wk = (const float*)d_in[5];
    const float* bk = (const float*)d_in[6];
    const float* Wv = (const float*)d_in[7];
    const float* bv = (const float*)d_in[8];
    const float* Wo = (const float*)d_in[9];
    const float* bo = (const float*)d_in[10];

    float* out    = (float*)d_out;                       // B*S*D
    float* attn_g = out + (size_t)MROWS * D_MODEL;       // B*H*S*S

    char* wsb = (char*)d_ws;
    const size_t act16 = (size_t)MROWS * D_MODEL * sizeof(_Float16); // 8 MB
    _Float16* qs = (_Float16*)(wsb);
    _Float16* ks = (_Float16*)(wsb + act16);
    _Float16* vs = (_Float16*)(wsb + 2 * act16);
    float* head_out = (float*)(wsb + 3 * act16);         // 16 MB fp32

    dim3 ggrid(MROWS / 16, D_MODEL / 64);
    gemm_bias_bT<_Float16><<<ggrid, 128, 0, stream>>>(q, Wq, bq, qs);
    gemm_bias_bT<_Float16><<<ggrid, 128, 0, stream>>>(k, Wk, bk, ks);
    gemm_bias_bT<_Float16><<<ggrid, 128, 0, stream>>>(v, Wv, bv, vs);

    attn_kernel<<<2 * NHEADS * (S_LEN / 16), 128, 0, stream>>>(qs, ks, vs, attn_g, head_out);

    gemm_bias_bT<float><<<ggrid, 128, 0, stream>>>(head_out, Wo, bo, out);
}
